// ConicaLayer_48430051230543
// MI455X (gfx1250) — matrix operand order repaired and checked by reading, not compiled, not run on hardware
//
#include <hip/hip_runtime.h>
#include <math.h>

// ---------------------------------------------------------------------------
// MI455X (gfx1250, wave32) transformer decoder layer:
//   self-attn (causal mask) -> post-LN -> cross-attn -> post-LN -> FFN(GELU) -> post-LN
// All matmuls via v_wmma_f32_16x16x32_bf16 (fp32 converted to bf16; f32 accum).
// LDS layouts arranged so every WMMA fragment is a contiguous b128-able read.
// ---------------------------------------------------------------------------

typedef __attribute__((ext_vector_type(16))) __bf16 v16bf;
typedef __attribute__((ext_vector_type(8)))  __bf16 v8bf;
typedef __attribute__((ext_vector_type(8)))  float  v8f;

#define B_   2
#define L_   2048
#define S_   1024
#define D_   1024
#define H_   16
#define DFF_ 4096
#define DH_  64

__device__ __forceinline__ __bf16 f2bf(float f) { return (__bf16)f; }

__device__ __forceinline__ v16bf cat8(v8bf lo, v8bf hi) {
  return __builtin_shufflevector(lo, hi, 0, 1, 2, 3, 4, 5, 6, 7,
                                 8, 9, 10, 11, 12, 13, 14, 15);
}

// Fragment layouts (ISA 7.12.2, wave32, 16-bit):
//  A 16x32 : lane m=lane&15, kh=lane>>4. code: dwords 0..3 = K kh*16+{0..7},
//            dwords 4..7 = K kh*16+8+{0..7}  -> two contiguous 16B chunks.
//  B 32x16 : lane n=lane&15, kh=lane>>4. code: dwords 0..7 = K kh*16+{0..15} as for A (not ISA order)
//            -> one contiguous 32B chunk when LDS is n-major.

// ---------------------------------------------------------------------------
// GEMM: C[M,N] = A[M,K] @ W[N,K]^T + bias[N], optional exact GELU.
// 256 threads = 8 waves. Tile 128x128, BK=32. Wave w owns rows w*16..w*16+15.
// ---------------------------------------------------------------------------
#define BM 128
#define BN 128
#define BK 32
#define BKP (BK + 8)   // 40 elems = 80 B row stride (16B aligned)

__global__ __launch_bounds__(256) void gemm_bias_kernel(
    const float* __restrict__ A, const float* __restrict__ W,
    const float* __restrict__ bias, float* __restrict__ C,
    int M, int N, int K, int act) {
  __shared__ __attribute__((aligned(16))) __bf16 As[BM][BKP];  // row-major (m,k)
  __shared__ __attribute__((aligned(16))) __bf16 Bs[BN][BKP];  // n-major  (n,k)

  const int tid  = threadIdx.x;
  const int lane = tid & 31;
  const int w    = tid >> 5;
  const int m0   = blockIdx.y * BM;
  const int n0   = blockIdx.x * BN;
  const int mh   = lane & 15;
  const int kh   = lane >> 4;

  v8f acc[8];
#pragma unroll
  for (int t = 0; t < 8; ++t)
#pragma unroll
    for (int e = 0; e < 8; ++e) acc[t][e] = 0.0f;

  const int sr = tid >> 1, scb = (tid & 1) * 16;  // staging: 16 elems/thread

  for (int k0 = 0; k0 < K; k0 += BK) {
    {  // A tile: global row-contiguous -> LDS row-contiguous
      const float4* src = (const float4*)(A + (size_t)(m0 + sr) * K + k0 + scb);
      v8bf p0, p1;
#pragma unroll
      for (int c4 = 0; c4 < 2; ++c4) {
        float4 x = src[c4];
        p0[c4 * 4 + 0] = f2bf(x.x); p0[c4 * 4 + 1] = f2bf(x.y);
        p0[c4 * 4 + 2] = f2bf(x.z); p0[c4 * 4 + 3] = f2bf(x.w);
      }
#pragma unroll
      for (int c4 = 0; c4 < 2; ++c4) {
        float4 x = src[2 + c4];
        p1[c4 * 4 + 0] = f2bf(x.x); p1[c4 * 4 + 1] = f2bf(x.y);
        p1[c4 * 4 + 2] = f2bf(x.z); p1[c4 * 4 + 3] = f2bf(x.w);
      }
      *(v8bf*)&As[sr][scb]     = p0;
      *(v8bf*)&As[sr][scb + 8] = p1;
    }
    {  // B tile: W row-contiguous -> LDS n-major row-contiguous
      const float4* src = (const float4*)(W + (size_t)(n0 + sr) * K + k0 + scb);
      v8bf p0, p1;
#pragma unroll
      for (int c4 = 0; c4 < 2; ++c4) {
        float4 x = src[c4];
        p0[c4 * 4 + 0] = f2bf(x.x); p0[c4 * 4 + 1] = f2bf(x.y);
        p0[c4 * 4 + 2] = f2bf(x.z); p0[c4 * 4 + 3] = f2bf(x.w);
      }
#pragma unroll
      for (int c4 = 0; c4 < 2; ++c4) {
        float4 x = src[2 + c4];
        p1[c4 * 4 + 0] = f2bf(x.x); p1[c4 * 4 + 1] = f2bf(x.y);
        p1[c4 * 4 + 2] = f2bf(x.z); p1[c4 * 4 + 3] = f2bf(x.w);
      }
      *(v8bf*)&Bs[sr][scb]     = p0;
      *(v8bf*)&Bs[sr][scb + 8] = p1;
    }
    if (k0 + BK < K) {  // global_prefetch_b8 of next tiles
      __builtin_prefetch(A + (size_t)(m0 + sr) * K + k0 + BK + scb, 0, 0);
      __builtin_prefetch(W + (size_t)(n0 + sr) * K + k0 + BK + scb, 0, 0);
    }
    __syncthreads();

    const int mrow = w * 16 + mh;
    v8bf alo = *(const v8bf*)&As[mrow][kh * 16];
    v8bf ahi = *(const v8bf*)&As[mrow][kh * 16 + 8];
    v16bf af = cat8(alo, ahi);
#pragma unroll
    for (int t = 0; t < 8; ++t) {
      v16bf bfr = *(const v16bf*)&Bs[t * 16 + mh][kh * 16];
      acc[t] = __builtin_amdgcn_wmma_f32_16x16x32_bf16(
          false, af, false, bfr, (short)0, acc[t], false, false);
    }
    __syncthreads();
  }

#pragma unroll
  for (int t = 0; t < 8; ++t) {
    const int col = n0 + t * 16 + mh;
    const float bv = bias ? bias[col] : 0.0f;
#pragma unroll
    for (int j = 0; j < 8; ++j) {
      const int row = m0 + w * 16 + j + kh * 8;
      if (row < M && col < N) {
        float vv = acc[t][j] + bv;
        if (act) vv = 0.5f * vv * (1.0f + erff(vv * 0.70710678118654752f));
        C[(size_t)row * N + col] = vv;
      }
    }
  }
}

// ---------------------------------------------------------------------------
// Flash attention: one block = (64 q-rows, one head, one batch). 4 waves.
// Q,K,V are [B*rows, D] with head h at cols h*64..h*64+63. mask is [L, T].
// Kt row-major (key,dh) -> contiguous B-frags for Q·K^T.
// Vtr transposed (dh,key) -> contiguous B-frags for P·V.
// ---------------------------------------------------------------------------
#define DHP (DH_ + 8)   // 72 elems = 144 B row stride (16B aligned)

__global__ __launch_bounds__(128) void attn_kernel(
    const float* __restrict__ Q, const float* __restrict__ K,
    const float* __restrict__ V, const float* __restrict__ mask,
    float* __restrict__ O, int Lq, int T, int D, float scale) {
  __shared__ __attribute__((aligned(16))) __bf16 Kt[64][DHP];
  __shared__ __attribute__((aligned(16))) __bf16 Vtr[DH_][DHP];
  __shared__ __attribute__((aligned(16))) __bf16 Pb[4][16][DHP];

  const int tid  = threadIdx.x;
  const int lane = tid & 31;
  const int w    = tid >> 5;
  const int b    = blockIdx.z;
  const int h    = blockIdx.y;
  const int q0   = blockIdx.x * 64 + w * 16;
  const int mh   = lane & 15;
  const int kh   = lane >> 4;
  const int dcol = h * DH_;

  // Q fragments (scaled), registers for the whole key loop
  v16bf qf[2];
  {
    const float* qp = Q + (size_t)(b * Lq + q0 + mh) * D + dcol;
#pragma unroll
    for (int ds = 0; ds < 2; ++ds) {
      v16bf qv;
#pragma unroll
      for (int e = 0; e < 8; ++e)
        qv[e] = f2bf(qp[ds * 32 + kh * 16 + e] * scale);
#pragma unroll
      for (int e = 0; e < 8; ++e)
        qv[8 + e] = f2bf(qp[ds * 32 + kh * 16 + 8 + e] * scale);
      qf[ds] = qv;
    }
  }

  v8f o_acc[4];
#pragma unroll
  for (int t = 0; t < 4; ++t)
#pragma unroll
    for (int e = 0; e < 8; ++e) o_acc[t][e] = 0.0f;
  float mrow[8], lrow[8];
#pragma unroll
  for (int j = 0; j < 8; ++j) { mrow[j] = -1e30f; lrow[j] = 0.0f; }

  const int cr = tid >> 1, ccb = (tid & 1) * 32;  // staging coords

  for (int t0 = 0; t0 < T; t0 += 64) {
    {
      const float4* kp = (const float4*)(K + (size_t)(b * T + t0 + cr) * D + dcol + ccb);
      const float4* vp = (const float4*)(V + (size_t)(b * T + t0 + cr) * D + dcol + ccb);
#pragma unroll
      for (int c4 = 0; c4 < 8; ++c4) {  // K: row-contiguous into Kt
        float4 x = kp[c4];
        v8bf p;
        // pack pairs of float4 into 16B chunks (two c4 per v8bf)
        p[0] = f2bf(x.x); p[1] = f2bf(x.y); p[2] = f2bf(x.z); p[3] = f2bf(x.w);
        float4 y = kp[++c4];
        p[4] = f2bf(y.x); p[5] = f2bf(y.y); p[6] = f2bf(y.z); p[7] = f2bf(y.w);
        *(v8bf*)&Kt[cr][ccb + (c4 - 1) * 4] = p;
      }
#pragma unroll
      for (int c4 = 0; c4 < 8; ++c4) {  // V: transpose into Vtr (dh-major)
        float4 x = vp[c4];
        const int dh = ccb + c4 * 4;
        Vtr[dh + 0][cr] = f2bf(x.x);
        Vtr[dh + 1][cr] = f2bf(x.y);
        Vtr[dh + 2][cr] = f2bf(x.z);
        Vtr[dh + 3][cr] = f2bf(x.w);
      }
    }
    __syncthreads();

    // S = Q K^T  (16 q-rows x 64 keys per wave)
    v8f s[4];
#pragma unroll
    for (int t = 0; t < 4; ++t)
#pragma unroll
      for (int e = 0; e < 8; ++e) s[t][e] = 0.0f;
#pragma unroll
    for (int ds = 0; ds < 2; ++ds)
#pragma unroll
      for (int t = 0; t < 4; ++t) {
        v16bf bfr = *(const v16bf*)&Kt[t * 16 + mh][ds * 32 + kh * 16];
        s[t] = __builtin_amdgcn_wmma_f32_16x16x32_bf16(
            false, qf[ds], false, bfr, (short)0, s[t], false, false);
      }

    // mask + online softmax (rows split across lane halves; xor<16 stays in half)
    float pv[4][8], alpha[8];
#pragma unroll
    for (int j = 0; j < 8; ++j) {
      const int qr = blockIdx.x * 64 + w * 16 + j + kh * 8;
      float mx = -1e30f;
#pragma unroll
      for (int t = 0; t < 4; ++t) {
        const int tc = t0 + t * 16 + mh;
        float sv = s[t][j] + mask[(size_t)qr * T + tc];
        s[t][j] = sv;
        mx = fmaxf(mx, sv);
      }
#pragma unroll
      for (int off = 1; off < 16; off <<= 1)
        mx = fmaxf(mx, __shfl_xor(mx, off, 32));
      const float mn = fmaxf(mrow[j], mx);
      alpha[j] = expf(mrow[j] - mn);
      mrow[j] = mn;
      float ps = 0.0f;
#pragma unroll
      for (int t = 0; t < 4; ++t) {
        float p = expf(s[t][j] - mn);
        pv[t][j] = p;
        ps += p;
      }
#pragma unroll
      for (int off = 1; off < 16; off <<= 1)
        ps += __shfl_xor(ps, off, 32);
      lrow[j] = lrow[j] * alpha[j] + ps;
    }

    // rescale O, spill P (C-layout) to per-wave LDS for A-fragment reload
#pragma unroll
    for (int t = 0; t < 4; ++t)
#pragma unroll
      for (int j = 0; j < 8; ++j) {
        o_acc[t][j] *= alpha[j];
        Pb[w][j + kh * 8][t * 16 + mh] = f2bf(pv[t][j]);
      }
    __syncthreads();

    // O += P V   (key dim is the WMMA K dim)
#pragma unroll
    for (int ds = 0; ds < 2; ++ds) {
      v8bf plo = *(const v8bf*)&Pb[w][mh][ds * 32 + kh * 16];
      v8bf phi = *(const v8bf*)&Pb[w][mh][ds * 32 + kh * 16 + 8];
      v16bf pf = cat8(plo, phi);
#pragma unroll
      for (int t = 0; t < 4; ++t) {
        v16bf vfr = *(const v16bf*)&Vtr[t * 16 + mh][ds * 32 + kh * 16];
        o_acc[t] = __builtin_amdgcn_wmma_f32_16x16x32_bf16(
            false, pf, false, vfr, (short)0, o_acc[t], false, false);
      }
    }
    __syncthreads();
  }

#pragma unroll
  for (int t = 0; t < 4; ++t)
#pragma unroll
    for (int j = 0; j < 8; ++j) {
      const int qr = q0 + j + kh * 8;
      const float denom = lrow[j] + 1e-20f;
      O[(size_t)(b * Lq + qr) * D + dcol + t * 16 + mh] = o_acc[t][j] / denom;
    }
}

// ---------------------------------------------------------------------------
// out[row] = LayerNorm(res[row] + x[row]) * g + b   (one block per row, D=1024)
// ---------------------------------------------------------------------------
__global__ __launch_bounds__(256) void addln_kernel(
    const float* __restrict__ res, const float* __restrict__ x,
    const float* __restrict__ g, const float* __restrict__ bta,
    float* __restrict__ out, int D) {
  __shared__ float red[256];
  const size_t row = blockIdx.x;
  const float* rp = res + row * D;
  const float* xp = x + row * D;
  float vals[4];
  float s = 0.0f;
#pragma unroll
  for (int i = 0; i < 4; ++i) {
    const int c = threadIdx.x + i * 256;
    vals[i] = rp[c] + xp[c];
    s += vals[i];
  }
  red[threadIdx.x] = s;
  __syncthreads();
  for (int o = 128; o > 0; o >>= 1) {
    if (threadIdx.x < o) red[threadIdx.x] += red[threadIdx.x + o];
    __syncthreads();
  }
  const float mean = red[0] / (float)D;
  __syncthreads();
  float vs = 0.0f;
#pragma unroll
  for (int i = 0; i < 4; ++i) {
    const float d = vals[i] - mean;
    vs += d * d;
  }
  red[threadIdx.x] = vs;
  __syncthreads();
  for (int o = 128; o > 0; o >>= 1) {
    if (threadIdx.x < o) red[threadIdx.x] += red[threadIdx.x + o];
    __syncthreads();
  }
  const float inv = rsqrtf(red[0] / (float)D + 1e-5f);
#pragma unroll
  for (int i = 0; i < 4; ++i) {
    const int c = threadIdx.x + i * 256;
    out[row * D + c] = (vals[i] - mean) * inv * g[c] + bta[c];
  }
}

// ---------------------------------------------------------------------------
extern "C" void kernel_launch(void* const* d_in, const int* in_sizes, int n_in,
                              void* d_out, int out_size, void* d_ws, size_t ws_size,
                              hipStream_t stream) {
  const float* hs    = (const float*)d_in[0];
  const float* enc   = (const float*)d_in[1];
  const float* amask = (const float*)d_in[2];
  const float* emask = (const float*)d_in[3];
  const float* sa_wq = (const float*)d_in[4];
  const float* sa_bq = (const float*)d_in[5];
  const float* sa_wk = (const float*)d_in[6];
  const float* sa_bk = (const float*)d_in[7];
  const float* sa_wv = (const float*)d_in[8];
  const float* sa_bv = (const float*)d_in[9];
  const float* sa_wo = (const float*)d_in[10];
  const float* sa_bo = (const float*)d_in[11];
  const float* sa_lg = (const float*)d_in[12];
  const float* sa_lb = (const float*)d_in[13];
  const float* ca_wq = (const float*)d_in[14];
  const float* ca_bq = (const float*)d_in[15];
  const float* ca_wk = (const float*)d_in[16];
  const float* ca_bk = (const float*)d_in[17];
  const float* ca_wv = (const float*)d_in[18];
  const float* ca_bv = (const float*)d_in[19];
  const float* ca_wo = (const float*)d_in[20];
  const float* ca_bo = (const float*)d_in[21];
  const float* ca_lg = (const float*)d_in[22];
  const float* ca_lb = (const float*)d_in[23];
  const float* f_w1  = (const float*)d_in[24];
  const float* f_b1  = (const float*)d_in[25];
  const float* f_w2  = (const float*)d_in[26];
  const float* f_b2  = (const float*)d_in[27];
  const float* f_lg  = (const float*)d_in[28];
  const float* f_lb  = (const float*)d_in[29];
  (void)in_sizes; (void)n_in; (void)out_size; (void)ws_size;

  const int ML = B_ * L_;   // 4096 query rows
  const int MS = B_ * S_;   // 2048 encoder rows
  const size_t szLD = (size_t)ML * D_;

  float* ws   = (float*)d_ws;
  float* q    = ws;
  float* k    = q + szLD;
  float* v    = k + szLD;
  float* attn = v + szLD;
  float* tmp  = attn + szLD;
  float* h    = tmp + szLD;
  float* ffn  = h + szLD;  // ML x DFF_

  const dim3 blk(256);
  const float scale = 0.125f;  // DH^-0.5 = 64^-0.5

  auto gg = [](int M, int N) { return dim3((N + BN - 1) / BN, (M + BM - 1) / BM); };

  // ---- self-attention block ----
  gemm_bias_kernel<<<gg(ML, D_), blk, 0, stream>>>(hs, sa_wq, sa_bq, q, ML, D_, D_, 0);
  gemm_bias_kernel<<<gg(ML, D_), blk, 0, stream>>>(hs, sa_wk, sa_bk, k, ML, D_, D_, 0);
  gemm_bias_kernel<<<gg(ML, D_), blk, 0, stream>>>(hs, sa_wv, sa_bv, v, ML, D_, D_, 0);
  attn_kernel<<<dim3(L_ / 64, H_, B_), 128, 0, stream>>>(q, k, v, amask, attn, L_, L_, D_, scale);
  gemm_bias_kernel<<<gg(ML, D_), blk, 0, stream>>>(attn, sa_wo, sa_bo, tmp, ML, D_, D_, 0);
  addln_kernel<<<ML, 256, 0, stream>>>(hs, tmp, sa_lg, sa_lb, h, D_);

  // ---- cross-attention block ----
  gemm_bias_kernel<<<gg(ML, D_), blk, 0, stream>>>(h, ca_wq, ca_bq, q, ML, D_, D_, 0);
  gemm_bias_kernel<<<gg(MS, D_), blk, 0, stream>>>(enc, ca_wk, ca_bk, k, MS, D_, D_, 0);
  gemm_bias_kernel<<<gg(MS, D_), blk, 0, stream>>>(enc, ca_wv, ca_bv, v, MS, D_, D_, 0);
  attn_kernel<<<dim3(L_ / 64, H_, B_), 128, 0, stream>>>(q, k, v, emask, attn, L_, S_, D_, scale);
  gemm_bias_kernel<<<gg(ML, D_), blk, 0, stream>>>(attn, ca_wo, ca_bo, tmp, ML, D_, D_, 0);
  addln_kernel<<<ML, 256, 0, stream>>>(h, tmp, ca_lg, ca_lb, h, D_);

  // ---- FFN block ----
  gemm_bias_kernel<<<gg(ML, DFF_), blk, 0, stream>>>(h, f_w1, f_b1, ffn, ML, DFF_, D_, 1);
  gemm_bias_kernel<<<gg(ML, D_), blk, 0, stream>>>(ffn, f_w2, f_b2, tmp, ML, D_, DFF_, 0);
  addln_kernel<<<ML, 256, 0, stream>>>(h, tmp, f_lg, f_lb, (float*)d_out, D_);
}
